// MultiHeadSelfAttention_6932077216235
// MI455X (gfx1250) — matrix operand order repaired and checked by reading, not compiled, not run on hardware
//
#include <hip/hip_runtime.h>
#include <math.h>
#include <stdint.h>

typedef __attribute__((ext_vector_type(16))) _Float16     v16h;
typedef __attribute__((ext_vector_type(8)))  _Float16     v8h;
typedef __attribute__((ext_vector_type(8)))  float        v8f;
typedef __attribute__((ext_vector_type(4)))  unsigned int v4u;
typedef __attribute__((ext_vector_type(8)))  int          v8i;
typedef __attribute__((ext_vector_type(4)))  int          v4i;

#define S   4096
#define DM  1024
#define H   16
#define DH  64

#if defined(__AMDGCN__)
#  if __has_builtin(__builtin_amdgcn_tensor_load_to_lds)
#    define HAVE_TDM 1
#  endif
#endif

union ABFrag { v16h v; v8h h[2]; };
union PairU32 { unsigned u; _Float16 f[2]; };

// Build a 16-half WMMA operand from two 16-byte chunks (A-frag and B^T-frag
// halves both [c..c+7],[c+16..c+23]).
static __device__ __forceinline__ v16h load2(const _Float16* p0, const _Float16* p1) {
    ABFrag u;
    u.h[0] = *(const v8h*)p0;
    u.h[1] = *(const v8h*)p1;
    return u.v;
}

#ifdef HAVE_TDM
// Issue one TDM 2-D tile load: tile_rows x tile_cols f16 elements, row stride
// stride_elems, packed row-major into LDS at l. D# packing per ISA 08 §8.
// This toolchain exposes the 6-arg builtin: (g0, g1, g2, g3, g4, cpol).
static __device__ __forceinline__ void tdm_load_2d(const _Float16* g, _Float16* l,
                                                   uint32_t tile_cols, uint32_t tile_rows,
                                                   uint32_t stride_elems) {
    uint64_t ga  = (uint64_t)(uintptr_t)g;
    uint32_t la  = (uint32_t)(uintptr_t)l;   // LDS byte addr = low 32 bits of generic ptr
    uint32_t td0 = tile_cols;                // tensor_dim0 (OOB bound, exact fit)
    uint32_t td1 = 1u << 20;                 // tensor_dim1: generous bound
    v4u g0;
    g0[0] = 1u;                                                 // count=1 user D#
    g0[1] = la;                                                 // lds_addr
    g0[2] = (uint32_t)ga;                                       // global_addr[31:0]
    g0[3] = (uint32_t)((ga >> 32) & 0x1FFFFFFu) | (2u << 30);   // addr[56:32] | type=2
    v8i g1;
    g1[0] = (int)(1u << 16);                                    // data_size = 2 bytes
    g1[1] = (int)((td0 & 0xFFFFu) << 16);                       // tensor_dim0[15:0]
    g1[2] = (int)((td0 >> 16) | ((td1 & 0xFFFFu) << 16));       // dim0 hi | dim1 lo
    g1[3] = (int)((td1 >> 16) | (tile_cols << 16));             // dim1 hi | tile_dim0
    g1[4] = (int)(tile_rows & 0xFFFFu);                         // tile_dim1, tile_dim2=0
    g1[5] = (int)stride_elems;                                  // tensor_dim0_stride lo
    g1[6] = 0;
    g1[7] = 0;
    v4i z4 = {0, 0, 0, 0};
    v8i z8 = {0, 0, 0, 0, 0, 0, 0, 0};
    __builtin_amdgcn_tensor_load_to_lds(g0, g1, z4, z4, z8, 0);
}
#endif

// Stage a rows x cols f16 tile (row stride ldg) into LDS, packed row-major.
// TDM path: one thread issues the DMA (EXEC ignored by TDM), overlap with
// compute, completed by stage_wait + block barrier. Fallback: cooperative copy.
template <int NT>
static __device__ __forceinline__ void stage_issue(const _Float16* g, _Float16* l,
                                                   int rows, int cols, int ldg, int tid) {
#ifdef HAVE_TDM
    if (tid == 0) tdm_load_2d(g, l, (uint32_t)cols, (uint32_t)rows, (uint32_t)ldg);
#else
    int chunks = rows * cols / 8;
    int cpr = cols / 8;
    for (int i = tid; i < chunks; i += NT) {
        int r = i / cpr, c = (i - r * cpr) * 8;
        *(v8h*)(l + r * cols + c) = *(const v8h*)(g + (size_t)r * ldg + c);
    }
#endif
}

static __device__ __forceinline__ void stage_wait(int tid) {
#ifdef HAVE_TDM
    if (tid == 0) asm volatile("s_wait_tensorcnt 0" ::: "memory");
#endif
}

// ---------------- Kernel 1: fp32 -> f16 conversion of x and weights ----------
__global__ void cvt_kernel(const float* __restrict__ x,
                           const float* __restrict__ wq, const float* __restrict__ wk,
                           const float* __restrict__ wv, const float* __restrict__ wo,
                           _Float16* __restrict__ xh,
                           _Float16* __restrict__ wqh, _Float16* __restrict__ wkh,
                           _Float16* __restrict__ wvh, _Float16* __restrict__ woh) {
    int i = blockIdx.x * 256 + threadIdx.x;
    int which = blockIdx.y;
    if (which == 0) {
        if (i < S * DM) xh[i] = (_Float16)x[i];
        return;
    }
    if (i >= DM * DM) return;
    const float* src = (which == 1) ? wq : (which == 2) ? wk : (which == 3) ? wv : wo;
    _Float16*    dst = (which == 1) ? wqh : (which == 2) ? wkh : (which == 3) ? wvh : woh;
    dst[i] = (_Float16)src[i];
}

// ---------------- Kernel 2: fused QKV projection GEMM (WMMA + TDM) ----------
// out[s,e] = sum_d x[s,d]*W[e,d]. W rows are exactly the B^T(NxK) layout.
// Per wave: 32x64 tile = 2 A-frags x 4 B-frags = 8 WMMAs per k-step; B frags
// loaded once from the TDM-staged LDS tile and reused across both A rows.
__global__ void qkv_kernel(const _Float16* __restrict__ xh,
                           const _Float16* __restrict__ wqh,
                           const _Float16* __restrict__ wkh,
                           const _Float16* __restrict__ wvh,
                           _Float16* __restrict__ qh,
                           _Float16* __restrict__ kh,
                           _Float16* __restrict__ vt) {
    __shared__ _Float16 bbuf[2][64 * 32];
    const int tid = threadIdx.x;
    const int lane = tid & 31, wave = tid >> 5;
    const int lo = lane & 15, hi = lane >> 4;
    const int m0 = blockIdx.x * 128 + wave * 32;
    const int n0 = blockIdx.y * 64;
    const int kind = blockIdx.z;  // 0=Q 1=K 2=V
    const _Float16* w = (kind == 0) ? wqh : (kind == 1) ? wkh : wvh;

    stage_issue<128>(w + (size_t)n0 * DM, bbuf[0], 64, 32, DM, tid);
    stage_wait(tid);
    __syncthreads();

    v8f c[2][4] = {};
    const _Float16* arow0 = xh + (size_t)(m0 + lo) * DM;
    const _Float16* arow1 = xh + (size_t)(m0 + 16 + lo) * DM;
    for (int k = 0; k < DM; k += 32) {
        int cur = (k >> 5) & 1;
        if (k + 32 < DM)  // prefetch next B tile into the other buffer
            stage_issue<128>(w + (size_t)n0 * DM + (k + 32), bbuf[cur ^ 1], 64, 32, DM, tid);
        v16h a0 = load2(arow0 + k + hi * 8, arow0 + k + 16 + hi * 8);
        v16h a1 = load2(arow1 + k + hi * 8, arow1 + k + 16 + hi * 8);
        const _Float16* bb = bbuf[cur];
        v16h bf[4];
#pragma unroll
        for (int nt = 0; nt < 4; ++nt) {
            const _Float16* brow = bb + (nt * 16 + lo) * 32 + hi * 8;
            bf[nt] = load2(brow, brow + 16);
        }
#pragma unroll
        for (int nt = 0; nt < 4; ++nt) {
            c[0][nt] = __builtin_amdgcn_wmma_f32_16x16x32_f16(false, a0, false, bf[nt],
                                                              (short)0, c[0][nt], false, false);
            c[1][nt] = __builtin_amdgcn_wmma_f32_16x16x32_f16(false, a1, false, bf[nt],
                                                              (short)0, c[1][nt], false, false);
        }
        stage_wait(tid);
        __syncthreads();
    }
    // Store epilogue: kind branch hoisted; V path packs (r,r+1) -> one b32
    // (consecutive r = consecutive s, contiguous in the [h][d][s] layout).
    if (kind == 2) {
#pragma unroll
        for (int mi = 0; mi < 2; ++mi) {
            int sbase = m0 + mi * 16 + 8 * hi;
            int e = n0 + lo;  // head-local column set below per nt
#pragma unroll
            for (int nt = 0; nt < 4; ++nt) {
                int ee = e + nt * 16;
                int hh = ee >> 6, d = ee & 63;
                _Float16* dst = vt + (size_t)(hh * DH + d) * S + sbase;
#pragma unroll
                for (int r = 0; r < 8; r += 2) {
                    PairU32 p;
                    p.f[0] = (_Float16)c[mi][nt][r];
                    p.f[1] = (_Float16)c[mi][nt][r + 1];
                    *(unsigned*)(dst + r) = p.u;
                }
            }
        }
    } else {
        _Float16* qk = (kind == 0) ? qh : kh;
#pragma unroll
        for (int mi = 0; mi < 2; ++mi) {
#pragma unroll
            for (int nt = 0; nt < 4; ++nt) {
                int e = n0 + nt * 16 + lo;
                int hh = e >> 6, d = e & 63;
#pragma unroll
                for (int r = 0; r < 8; ++r) {
                    int s = m0 + mi * 16 + r + 8 * hi;
                    qk[((size_t)hh * S + s) * DH + d] = (_Float16)c[mi][nt][r];
                }
            }
        }
    }
}

// ---------------- Kernel 3: RoPE in place on Q and K -------------------------
// One thread owns one aligned (even,odd) pair: single b32 load/store per tensor.
__global__ void rope_kernel(const int* __restrict__ tpos,
                            _Float16* __restrict__ qh, _Float16* __restrict__ kh) {
    int idx = blockIdx.x * 256 + threadIdx.x;  // one thread per (h,s,pair)
    int i = idx & 31;
    int s = (idx >> 5) & (S - 1);
    int h = idx >> 17;
    float pos  = (float)tpos[s];
    float freq = __powf(10000.0f, -(float)(2 * i) / 64.0f);
    float ang  = pos * freq;
    float cs = __cosf(ang), sn = __sinf(ang);
    size_t base = ((size_t)h * S + s) * DH + 2 * i;
    unsigned* qp = (unsigned*)(qh + base);
    unsigned* kp = (unsigned*)(kh + base);
    PairU32 q, k2;
    q.u  = *qp;
    k2.u = *kp;
    float qe = (float)q.f[0], qo = (float)q.f[1];
    q.f[0] = (_Float16)(qe * cs - qo * sn);
    q.f[1] = (_Float16)(qe * sn + qo * cs);
    *qp = q.u;
    float ke = (float)k2.f[0], ko = (float)k2.f[1];
    k2.f[0] = (_Float16)(ke * cs - ko * sn);
    k2.f[1] = (_Float16)(ke * sn + ko * cs);
    *kp = k2.u;
}

// ---------------- Kernel 4: causal flash attention (WMMA + TDM staging) ------
// Block = 8 waves x 16 query rows = 128 rows. K/V 32-key tiles double-buffered
// in LDS by TDM (shared by all 8 waves). All waves run the block-uniform chunk
// count (barrier-legal); each wave gates its own compute by its causal limit.
// Fragment loads are hoisted ahead of each WMMA chain; V frags are loaded
// before the softmax VALU block so LDS latency hides under it.
__global__ void attn_kernel(const _Float16* __restrict__ qh,
                            const _Float16* __restrict__ kh,
                            const _Float16* __restrict__ vt,
                            _Float16* __restrict__ attn) {
    __shared__ _Float16 kb[2][32 * 64];   // 32 key rows x 64 d
    __shared__ _Float16 vb[2][64 * 32];   // 64 d rows x 32 keys (V^T)
    __shared__ _Float16 ptile[8][16 * 32];
    const int tid = threadIdx.x;
    const int lane = tid & 31, wave = tid >> 5;
    const int lo = lane & 15, hi = lane >> 4;
    const int h  = blockIdx.y;
    const int m0 = blockIdx.x * 128 + wave * 16;
    _Float16* pt = ptile[wave];

    const _Float16* qr = qh + ((size_t)h * S + m0 + lo) * DH;
    v16h aq0 = load2(qr + hi * 8,      qr + 16 + hi * 8);
    v16h aq1 = load2(qr + 32 + hi * 8, qr + 48 + hi * 8);

    float m[8], l[8];
    v8f o[4] = {};
#pragma unroll
    for (int r = 0; r < 8; ++r) { m[r] = -INFINITY; l[r] = 0.0f; }

    const float scale = 0.125f;  // 1/sqrt(64)
    const int mlast   = m0 + 15;
    const int nchunks = blockIdx.x * 4 + 4;  // block-uniform causal chunk count

    stage_issue<256>(kh + (size_t)h * S * DH, kb[0], 32, 64, DH, tid);
    stage_issue<256>(vt + (size_t)h * DH * S, vb[0], 64, 32, S, tid);
    stage_wait(tid);
    __syncthreads();

    for (int cc = 0; cc < nchunks; ++cc) {
        const int jc  = cc * 32;
        const int cur = cc & 1;
        if (cc + 1 < nchunks) {  // prefetch next K/V tiles
            stage_issue<256>(kh + ((size_t)h * S + jc + 32) * DH, kb[cur ^ 1], 32, 64, DH, tid);
            stage_issue<256>(vt + (size_t)h * DH * S + jc + 32,   vb[cur ^ 1], 64, 32, S, tid);
        }
        if (jc <= mlast) {  // wave-uniform causal gate
            // ---- hoist all K fragments, then run the 4-WMMA score chain
            v16h kf0[2], kf1[2];
#pragma unroll
            for (int ct = 0; ct < 2; ++ct) {
                const _Float16* krow = kb[cur] + (ct * 16 + lo) * 64 + hi * 8;
                kf0[ct] = load2(krow,      krow + 16);
                kf1[ct] = load2(krow + 32, krow + 48);
            }
            v8f sc[2];
#pragma unroll
            for (int ct = 0; ct < 2; ++ct) {
                v8f z = {};
                z      = __builtin_amdgcn_wmma_f32_16x16x32_f16(false, aq0, false, kf0[ct],
                                                                (short)0, z, false, false);
                sc[ct] = __builtin_amdgcn_wmma_f32_16x16x32_f16(false, aq1, false, kf1[ct],
                                                                (short)0, z, false, false);
            }
            // ---- V fragments issued now; LDS latency hides under softmax VALU
            v16h vf[4];
#pragma unroll
            for (int nt = 0; nt < 4; ++nt) {
                const _Float16* vrow = vb[cur] + (nt * 16 + lo) * 32 + hi * 8;
                vf[nt] = load2(vrow, vrow + 16);
            }
            // ---- causal mask + scale + online softmax (row M = r + 8*hi)
#pragma unroll
            for (int r = 0; r < 8; ++r) {
                int qrow = m0 + r + 8 * hi;
#pragma unroll
                for (int ct = 0; ct < 2; ++ct) {
                    int jcol = jc + ct * 16 + lo;
                    float v = sc[ct][r] * scale;
                    sc[ct][r] = (jcol <= qrow) ? v : -INFINITY;
                }
                float mm = fmaxf(sc[0][r], sc[1][r]);
                mm = fmaxf(mm, __shfl_xor(mm, 1));
                mm = fmaxf(mm, __shfl_xor(mm, 2));
                mm = fmaxf(mm, __shfl_xor(mm, 4));
                mm = fmaxf(mm, __shfl_xor(mm, 8));
                float mnew  = fmaxf(m[r], mm);
                float alpha = __expf(m[r] - mnew);
                float p0 = __expf(sc[0][r] - mnew);
                float p1 = __expf(sc[1][r] - mnew);
                float rs = p0 + p1;
                rs += __shfl_xor(rs, 1);
                rs += __shfl_xor(rs, 2);
                rs += __shfl_xor(rs, 4);
                rs += __shfl_xor(rs, 8);
                l[r] = l[r] * alpha + rs;
                m[r] = mnew;
#pragma unroll
                for (int nt = 0; nt < 4; ++nt) o[nt][r] *= alpha;
                pt[(r + 8 * hi) * 32 + lo]      = (_Float16)p0;
                pt[(r + 8 * hi) * 32 + 16 + lo] = (_Float16)p1;
            }
            // per-wave P-tile visibility (no block barrier needed)
            asm volatile("s_wait_dscnt 0" ::: "memory");
            __builtin_amdgcn_wave_barrier();
            v16h pa = load2(pt + lo * 32 + hi * 8, pt + lo * 32 + 16 + hi * 8);
            __builtin_amdgcn_wave_barrier();
            // ---- PV: 4 back-to-back WMMAs on preloaded V frags
#pragma unroll
            for (int nt = 0; nt < 4; ++nt) {
                o[nt] = __builtin_amdgcn_wmma_f32_16x16x32_f16(false, pa, false, vf[nt],
                                                               (short)0, o[nt], false, false);
            }
        }
        stage_wait(tid);
        __syncthreads();
    }
    // ---- normalize and store attn[s][e] (e = h*64 + d), f16 for out-proj
#pragma unroll
    for (int nt = 0; nt < 4; ++nt) {
#pragma unroll
        for (int r = 0; r < 8; ++r) {
            int srow = m0 + r + 8 * hi;
            int e = h * DH + nt * 16 + lo;
            float inv = (l[r] > 0.0f) ? 1.0f / l[r] : 0.0f;
            attn[(size_t)srow * DM + e] = (_Float16)(o[nt][r] * inv);
        }
    }
}

// ---------------- Kernel 5: output projection GEMM (WMMA + TDM), f32 out -----
__global__ void oproj_kernel(const _Float16* __restrict__ attn,
                             const _Float16* __restrict__ woh,
                             float* __restrict__ out) {
    __shared__ _Float16 bbuf[2][64 * 32];
    const int tid = threadIdx.x;
    const int lane = tid & 31, wave = tid >> 5;
    const int lo = lane & 15, hi = lane >> 4;
    const int m0 = blockIdx.x * 128 + wave * 32;
    const int n0 = blockIdx.y * 64;

    stage_issue<128>(woh + (size_t)n0 * DM, bbuf[0], 64, 32, DM, tid);
    stage_wait(tid);
    __syncthreads();

    v8f c[2][4] = {};
    const _Float16* arow0 = attn + (size_t)(m0 + lo) * DM;
    const _Float16* arow1 = attn + (size_t)(m0 + 16 + lo) * DM;
    for (int k = 0; k < DM; k += 32) {
        int cur = (k >> 5) & 1;
        if (k + 32 < DM)
            stage_issue<128>(woh + (size_t)n0 * DM + (k + 32), bbuf[cur ^ 1], 64, 32, DM, tid);
        v16h a0 = load2(arow0 + k + hi * 8, arow0 + k + 16 + hi * 8);
        v16h a1 = load2(arow1 + k + hi * 8, arow1 + k + 16 + hi * 8);
        const _Float16* bb = bbuf[cur];
        v16h bf[4];
#pragma unroll
        for (int nt = 0; nt < 4; ++nt) {
            const _Float16* brow = bb + (nt * 16 + lo) * 32 + hi * 8;
            bf[nt] = load2(brow, brow + 16);
        }
#pragma unroll
        for (int nt = 0; nt < 4; ++nt) {
            c[0][nt] = __builtin_amdgcn_wmma_f32_16x16x32_f16(false, a0, false, bf[nt],
                                                              (short)0, c[0][nt], false, false);
            c[1][nt] = __builtin_amdgcn_wmma_f32_16x16x32_f16(false, a1, false, bf[nt],
                                                              (short)0, c[1][nt], false, false);
        }
        stage_wait(tid);
        __syncthreads();
    }
#pragma unroll
    for (int mi = 0; mi < 2; ++mi)
#pragma unroll
        for (int nt = 0; nt < 4; ++nt)
#pragma unroll
            for (int r = 0; r < 8; ++r)
                out[(size_t)(m0 + mi * 16 + r + 8 * hi) * DM + n0 + nt * 16 + lo] = c[mi][nt][r];
}

extern "C" void kernel_launch(void* const* d_in, const int* in_sizes, int n_in,
                              void* d_out, int out_size, void* d_ws, size_t ws_size,
                              hipStream_t stream) {
    const float* x  = (const float*)d_in[0];
    const int*   tp = (const int*)d_in[1];
    const float* wq = (const float*)d_in[2];
    const float* wk = (const float*)d_in[3];
    const float* wv = (const float*)d_in[4];
    const float* wo = (const float*)d_in[5];
    float* out = (float*)d_out;

    // workspace layout (f16): ~48 MB total
    _Float16* ws = (_Float16*)d_ws;
    size_t off = 0;
    _Float16* xh   = ws + off; off += (size_t)S * DM;
    _Float16* wqh  = ws + off; off += (size_t)DM * DM;
    _Float16* wkh  = ws + off; off += (size_t)DM * DM;
    _Float16* wvh  = ws + off; off += (size_t)DM * DM;
    _Float16* woh  = ws + off; off += (size_t)DM * DM;
    _Float16* qh   = ws + off; off += (size_t)S * DM;   // [h][s][d]
    _Float16* kh   = ws + off; off += (size_t)S * DM;   // [h][s][d]
    _Float16* vt   = ws + off; off += (size_t)S * DM;   // [h][d][s]
    _Float16* attn = ws + off; off += (size_t)S * DM;   // [s][e]

    cvt_kernel<<<dim3(16384, 5), 256, 0, stream>>>(x, wq, wk, wv, wo,
                                                   xh, wqh, wkh, wvh, woh);
    qkv_kernel<<<dim3(S / 128, DM / 64, 3), 128, 0, stream>>>(xh, wqh, wkh, wvh,
                                                              qh, kh, vt);
    rope_kernel<<<dim3((H * S * (DH / 2)) / 256), 256, 0, stream>>>(tp, qh, kh);
    attn_kernel<<<dim3(S / 128, H), 256, 0, stream>>>(qh, kh, vt, attn);
    oproj_kernel<<<dim3(S / 128, DM / 64), 128, 0, stream>>>(attn, woh, out);
}
